// SelfModel_76630806495739
// MI455X (gfx1250) — matrix operand order repaired and checked by reading, not compiled, not run on hardware
//
#include <hip/hip_runtime.h>
#include <hip/hip_bf16.h>
#include <math.h>

typedef _Float16 half_t;
typedef __attribute__((ext_vector_type(16))) _Float16 v16h;
typedef __attribute__((ext_vector_type(8)))  float    v8f;

#define DEV static __device__ __forceinline__
#define WMMA(acc, a, b) \
  __builtin_amdgcn_wmma_f32_16x16x32_f16(false, (a), false, (b), (short)0, (acc), false, false)

// ---------------- problem constants ----------------
constexpr int Bb = 4, Dd = 14, Nn = 1024, Ff = 64;
constexpr int HID = 128, OUTC = 64, H1n = 128, H2n = 64, Yn = 7;
constexpr int BD = Bb * Dd;            // 56
constexpr int NODES = Bb * Nn;         // 4096

// ---------------- workspace layout ----------------
constexpr size_t alignup(size_t x) { return (x + 255) & ~size_t(255); }
constexpr size_t OFF_W1T   = 0;                                              // [128][64] f16
constexpr size_t OFF_W2T   = alignup(OFF_W1T   + 128 * 64 * 2);              // [64][128] f16
constexpr size_t OFF_WC1   = alignup(OFF_W2T   + 64 * 128 * 2);              // [512][256] f16
constexpr size_t OFF_WIH2T = alignup(OFF_WC1   + 512 * 256 * 2);             // [256][128] f16
constexpr size_t OFF_BC1   = alignup(OFF_WIH2T + 256 * 128 * 2);             // [512] f32
constexpr size_t OFF_BC2   = alignup(OFF_BC1   + 512 * 4);                   // [256] f32
constexpr size_t OFF_DINV  = alignup(OFF_BC2   + 256 * 4);                   // [56*1024] f32
constexpr size_t OFF_XW1T  = alignup(OFF_DINV  + (size_t)BD * Nn * 4);       // [56][128][1024] f16
constexpr size_t OFF_HF    = alignup(OFF_XW1T  + (size_t)BD * 128 * Nn * 2); // [56][1024][128] f16
constexpr size_t OFF_HW2T  = alignup(OFF_HF    + (size_t)BD * Nn * 128 * 2); // [56][64][1024] f16
constexpr size_t OFF_Z     = alignup(OFF_HW2T  + (size_t)BD * 64 * Nn * 2);  // [56][1024][128] f16
constexpr size_t OFF_H1    = alignup(OFF_Z     + (size_t)BD * Nn * 128 * 2); // [4096][128] f16
constexpr size_t OFF_C1    = alignup(OFF_H1    + (size_t)NODES * 128 * 2);   // [4096][128] f32
constexpr size_t OFF_H2    = alignup(OFF_C1    + (size_t)NODES * 128 * 4);   // [4096][64]  f32

// ---------------- WMMA fragment helpers (ISA 7.12.2 layouts) ----------------
// A 16x32 f16, row-major source (fp32): lane<16 -> rows m0+lane, K chunks {0..7},{16..23};
// lanes 16..31 same rows, +8 K offset.
DEV v16h load_a_f32(const float* src, int ld, int m0, int k0, int lane) {
  const int hf = lane >> 4;
  const float* row = src + (size_t)(m0 + (lane & 15)) * ld;
  const int kb0 = k0 + 8 * hf, kb1 = k0 + 16 + 8 * hf;
  v16h a;
#pragma unroll
  for (int e = 0; e < 8; ++e) {
    a[e]     = (half_t)row[kb0 + e];
    a[8 + e] = (half_t)row[kb1 + e];
  }
  return a;
}

DEV v16h load_a_f16(const half_t* src, int ld, int m0, int k0, int lane) {
  const int hf = lane >> 4;
  const half_t* row = src + (size_t)(m0 + (lane & 15)) * ld;
  const int kb0 = k0 + 8 * hf, kb1 = k0 + 16 + 8 * hf;
  v16h a;
#pragma unroll
  for (int e = 0; e < 8; ++e) {
    a[e]     = row[kb0 + e];
    a[8 + e] = row[kb1 + e];
  }
  return a;
}

// A fragment of normA = dinv_i * (A_ij + I) * dinv_j, built on the fly from fp32 A.
DEV v16h load_a_norm(const float* Arow, const float* s_dinv, float di, int i, int k0, int lane) {
  const int hf = lane >> 4;
  const int kb0 = k0 + 8 * hf, kb1 = k0 + 16 + 8 * hf;
  v16h a;
#pragma unroll
  for (int e = 0; e < 8; ++e) {
    const int j0 = kb0 + e, j1 = kb1 + e;
    const float v0 = Arow[j0] + (j0 == i ? 1.0f : 0.0f);
    const float v1 = Arow[j1] + (j1 == i ? 1.0f : 0.0f);
    a[e]     = (half_t)(di * v0 * s_dinv[j0]);
    a[8 + e] = (half_t)(di * v1 * s_dinv[j1]);
  }
  return a;
}

// B 32x16 f16 from transposed storage Bt[N][K]: K chunks {0..7},{16..23} (+8 for lanes 16..31).
DEV v16h load_b_t(const half_t* Bt, int ldb, int n0, int k0, int lane) {
  const half_t* p = Bt + (size_t)(n0 + (lane & 15)) * ldb + k0 + 8 * (lane >> 4);
  v16h b;
#pragma unroll
  for (int e = 0; e < 8; ++e) { b[e] = p[e]; b[8 + e] = p[16 + e]; }
  return b;
}

// C 16x16 f32 tile stored TRANSPOSED as f16 [N][ldm]: contiguous 8-element store per lane.
DEV void store_c_t16(half_t* Ct, int ldm, int m0, int n0, const v8f& acc, int lane) {
  half_t* p = Ct + (size_t)(n0 + (lane & 15)) * ldm + m0 + 8 * (lane >> 4);
#pragma unroll
  for (int r = 0; r < 8; ++r) p[r] = (half_t)acc[r];
}

DEV float sigm(float x) { return 1.0f / (1.0f + expf(-x)); }

// ---------------- kernels ----------------

// Repack weights to f16 transposed operands; combine biases. One block.
__global__ void prep_weights_kernel(const float* __restrict__ w1, const float* __restrict__ w2,
                                    const float* __restrict__ wih1, const float* __restrict__ whh1,
                                    const float* __restrict__ bih1, const float* __restrict__ bhh1,
                                    const float* __restrict__ wih2, const float* __restrict__ bih2,
                                    const float* __restrict__ bhh2,
                                    half_t* __restrict__ w1t, half_t* __restrict__ w2t,
                                    half_t* __restrict__ Wc1, half_t* __restrict__ wih2t,
                                    float* __restrict__ bc1, float* __restrict__ bc2) {
  const int tid = threadIdx.x;
  for (int i = tid; i < 128 * 64; i += 256) { int h = i / 64, k = i % 64;  w1t[i] = (half_t)w1[k * 128 + h]; }
  for (int i = tid; i < 64 * 128; i += 256) { int o = i / 128, k = i % 128; w2t[i] = (half_t)w2[k * 64 + o]; }
  for (int i = tid; i < 512 * 256; i += 256) {
    int g = i / 256, k = i % 256;
    Wc1[i] = (half_t)(k < 128 ? wih1[k * 512 + g] : whh1[(k - 128) * 512 + g]);
  }
  for (int i = tid; i < 256 * 128; i += 256) { int g = i / 128, k = i % 128; wih2t[i] = (half_t)wih2[k * 256 + g]; }
  for (int i = tid; i < 512; i += 256) bc1[i] = bih1[i] + bhh1[i];
  for (int i = tid; i < 256; i += 256) bc2[i] = bih2[i] + bhh2[i];
}

__global__ void zero_state_kernel(half_t* __restrict__ H1, float* __restrict__ C1) {
  const int i = blockIdx.x * 256 + threadIdx.x;
  if (i < NODES * 128) { H1[i] = (half_t)0.0f; C1[i] = 0.0f; }
}

// dinv = rsqrt(rowsum(A) + 1); one wave per row.
__global__ void degree_kernel(const float* __restrict__ A, float* __restrict__ dinv) {
  const int wave = threadIdx.x >> 5, lane = threadIdx.x & 31;
  const int row = blockIdx.x * 8 + wave;
  const float* Ar = A + (size_t)row * Nn;
  float s = 0.0f;
  for (int j = lane; j < Nn; j += 32) s += Ar[j];
#pragma unroll
  for (int off = 16; off > 0; off >>= 1) s += __shfl_xor(s, off, 32);
  if (lane == 0) dinv[row] = rsqrtf(s + 1.0f);
}

// XW1t[bd][128][1024] = (X @ w1)^T  (f16, transposed -> B operand of GCN1)
__global__ void xw1_kernel(const float* __restrict__ X, const half_t* __restrict__ w1t,
                           half_t* __restrict__ XW1T) {
  const int bd = blockIdx.x >> 3, rb = blockIdx.x & 7;
  const int wave = threadIdx.x >> 5, lane = threadIdx.x & 31;
  const int m0 = rb * 128 + wave * 16;
  const float* Xbd = X + (size_t)bd * Nn * Ff;
  v8f acc[8] = {};
  for (int k0 = 0; k0 < Ff; k0 += 32) {
    v16h a = load_a_f32(Xbd, Ff, m0, k0, lane);
#pragma unroll
    for (int t = 0; t < 8; ++t) {
      v16h b = load_b_t(w1t, 64, t * 16, k0, lane);
      acc[t] = WMMA(acc[t], a, b);
    }
  }
  half_t* Obd = XW1T + (size_t)bd * 128 * Nn;
#pragma unroll
  for (int t = 0; t < 8; ++t) store_c_t16(Obd, Nn, m0, t * 16, acc[t], lane);
}

// out = [relu](normA @ B + bias); normalization fused into the A-fragment load.
// Writes row-major f16 [bd][node][out_ld] (columns 0..COLS-1).
template <int COLS, bool RELU>
__global__ void gcn_kernel(const float* __restrict__ A, const float* __restrict__ dinv,
                           const half_t* __restrict__ Bt, const float* __restrict__ bias,
                           half_t* __restrict__ Out, int out_ld) {
  __shared__ float s_dinv[Nn];
  const int bd = blockIdx.x >> 3, rb = blockIdx.x & 7;
  for (int i = threadIdx.x; i < Nn; i += 256) s_dinv[i] = dinv[bd * Nn + i];
  __syncthreads();
  const int wave = threadIdx.x >> 5, lane = threadIdx.x & 31;
  const int m0 = rb * 128 + wave * 16;
  const int i = m0 + (lane & 15);
  const float di = s_dinv[i];
  const float* Arow = A + ((size_t)bd * Nn + i) * Nn;
  const half_t* Bbd = Bt + (size_t)bd * COLS * Nn;
  constexpr int NT = COLS / 16;
  v8f acc[NT] = {};
  for (int k0 = 0; k0 < Nn; k0 += 32) {
    v16h a = load_a_norm(Arow, s_dinv, di, i, k0, lane);
#pragma unroll
    for (int t = 0; t < NT; ++t) {
      v16h b = load_b_t(Bbd, Nn, t * 16, k0, lane);
      acc[t] = WMMA(acc[t], a, b);
    }
  }
  half_t* Orow = Out + (size_t)bd * Nn * out_ld;
  const int nlo = lane & 15, mb = m0 + 8 * (lane >> 4);
#pragma unroll
  for (int t = 0; t < NT; ++t) {
    const float bv = bias[t * 16 + nlo];
#pragma unroll
    for (int r = 0; r < 8; ++r) {
      float v = acc[t][r] + bv;
      if (RELU) v = fmaxf(v, 0.0f);
      Orow[(size_t)(mb + r) * out_ld + t * 16 + nlo] = (half_t)v;
    }
  }
}

// HW2t[bd][64][1024] = (h @ w2)^T  (B operand of GCN2)
__global__ void hw2_kernel(const half_t* __restrict__ H, const half_t* __restrict__ w2t,
                           half_t* __restrict__ HW2T) {
  const int bd = blockIdx.x >> 3, rb = blockIdx.x & 7;
  const int wave = threadIdx.x >> 5, lane = threadIdx.x & 31;
  const int m0 = rb * 128 + wave * 16;
  const half_t* Hbd = H + (size_t)bd * Nn * HID;
  v8f acc[4] = {};
  for (int k0 = 0; k0 < HID; k0 += 32) {
    v16h a = load_a_f16(Hbd, HID, m0, k0, lane);
#pragma unroll
    for (int t = 0; t < 4; ++t) {
      v16h b = load_b_t(w2t, HID, t * 16, k0, lane);
      acc[t] = WMMA(acc[t], a, b);
    }
  }
  half_t* Obd = HW2T + (size_t)bd * 64 * Nn;
#pragma unroll
  for (int t = 0; t < 4; ++t) store_c_t16(Obd, Nn, m0, t * 16, acc[t], lane);
}

// Text branch: z[...,64+o] = (extra @ wt + bt) * exp(idx_b * lam_n^2)
__global__ void text_kernel(const float* __restrict__ extra, const float* __restrict__ idxv,
                            const float* __restrict__ wt, const float* __restrict__ bt,
                            const float* __restrict__ lam, half_t* __restrict__ Z) {
  const size_t i = (size_t)blockIdx.x * 256 + threadIdx.x;
  if (i >= (size_t)BD * Nn * OUTC) return;
  const int o = (int)(i & 63);
  const size_t bdn = i >> 6;                 // bd*1024 + n
  const int n = (int)(bdn & 1023);
  const int bd = (int)(bdn >> 10);
  const int b = bd / Dd;
  const float* e = extra + bdn * 32;
  float s = bt[o];
#pragma unroll
  for (int k = 0; k < 32; ++k) s += e[k] * wt[k * 64 + o];
  const float sc = expf(idxv[b] * lam[n] * lam[n]);
  Z[bdn * 128 + 64 + o] = (half_t)(s * sc);
}

// One LSTM1 time step: gates = [x_t | h] @ [Wih;Whh] + (bih+bhh), fused cell update.
// Block = 16 node rows x all 512 gate cols (8 waves, one 16-col slot each x 4 gates).
__global__ void lstm1_step_kernel(const half_t* __restrict__ Z, const half_t* __restrict__ Wc1,
                                  const float* __restrict__ bc1, half_t* __restrict__ H1,
                                  float* __restrict__ C1, int t) {
  const int wave = threadIdx.x >> 5, lane = threadIdx.x & 31;
  const int m0 = blockIdx.x * 16;            // node tile
  const int n0 = wave * 16;                  // slot within H1 width
  const int bb = m0 >> 10;                   // batch index (tile never straddles batches)
  const half_t* xbase = Z + (((size_t)bb * Dd + t) * Nn + (m0 & 1023)) * 128;
  v8f acc[4] = {};
  for (int k0 = 0; k0 < 256; k0 += 32) {
    v16h a = (k0 < 128) ? load_a_f16(xbase, 128, 0, k0, lane)
                        : load_a_f16(H1 + (size_t)m0 * 128, 128, 0, k0 - 128, lane);
#pragma unroll
    for (int g = 0; g < 4; ++g) {
      v16h b = load_b_t(Wc1, 256, n0 + g * 128, k0, lane);
      acc[g] = WMMA(acc[g], a, b);
    }
  }
  __syncthreads();  // all waves finished reading this tile's H1 rows before overwrite
  const int cn = n0 + (lane & 15);
  const int mb = m0 + 8 * (lane >> 4);
  const float bi = bc1[cn], bff = bc1[cn + 128], bg = bc1[cn + 256], bo = bc1[cn + 384];
#pragma unroll
  for (int r = 0; r < 8; ++r) {
    const size_t off = (size_t)(mb + r) * 128 + cn;
    const float iv = sigm(acc[0][r] + bi);
    const float fv = sigm(acc[1][r] + bff);
    const float gv = tanhf(acc[2][r] + bg);
    const float ov = sigm(acc[3][r] + bo);
    const float cnew = fv * C1[off] + iv * gv;
    C1[off] = cnew;
    H1[off] = (half_t)(ov * tanhf(cnew));
  }
}

// LSTM2: single cell from zero state: gates = h1 @ wih2 + (bih2+bhh2)
__global__ void lstm2_kernel(const half_t* __restrict__ H1, const half_t* __restrict__ wih2t,
                             const float* __restrict__ bc2, float* __restrict__ H2) {
  const int wave = threadIdx.x >> 5, lane = threadIdx.x & 31;
  const int m0 = (blockIdx.x * 2 + (wave >> 2)) * 16;
  const int n0 = (wave & 3) * 16;
  v8f acc[4] = {};
  for (int k0 = 0; k0 < 128; k0 += 32) {
    v16h a = load_a_f16(H1, 128, m0, k0, lane);
#pragma unroll
    for (int g = 0; g < 4; ++g) {
      v16h b = load_b_t(wih2t, 128, n0 + g * 64, k0, lane);
      acc[g] = WMMA(acc[g], a, b);
    }
  }
  const int cn = n0 + (lane & 15);
  const int mb = m0 + 8 * (lane >> 4);
  const float bi = bc2[cn], bg = bc2[cn + 128], bo = bc2[cn + 192];
#pragma unroll
  for (int r = 0; r < 8; ++r) {
    const float cv = sigm(acc[0][r] + bi) * tanhf(acc[2][r] + bg);
    H2[(size_t)(mb + r) * 64 + cn] = sigm(acc[3][r] + bo) * tanhf(cv);
  }
}

// out[bn*7+y] = h2[bn] @ wf + bf  (raw reshape of [4096,7] matches d_out layout)
__global__ void final_kernel(const float* __restrict__ H2, const float* __restrict__ wf,
                             const float* __restrict__ bfv, float* __restrict__ out) {
  const int i = blockIdx.x * 256 + threadIdx.x;
  if (i >= NODES * Yn) return;
  const int bn = i / Yn, y = i % Yn;
  const float* h = H2 + (size_t)bn * 64;
  float s = bfv[y];
#pragma unroll
  for (int k = 0; k < 64; ++k) s += h[k] * wf[k * Yn + y];
  out[i] = s;
}

// ---------------- launcher ----------------
extern "C" void kernel_launch(void* const* d_in, const int* in_sizes, int n_in,
                              void* d_out, int out_size, void* d_ws, size_t ws_size,
                              hipStream_t stream) {
  (void)in_sizes; (void)n_in; (void)out_size; (void)ws_size;
  const float* X     = (const float*)d_in[0];
  const float* A     = (const float*)d_in[1];
  const float* extra = (const float*)d_in[2];
  const float* idxv  = (const float*)d_in[3];
  const float* w1    = (const float*)d_in[4];
  const float* b1    = (const float*)d_in[5];
  const float* w2    = (const float*)d_in[6];
  const float* b2    = (const float*)d_in[7];
  const float* wt    = (const float*)d_in[8];
  const float* bt    = (const float*)d_in[9];
  const float* wih1  = (const float*)d_in[10];
  const float* whh1  = (const float*)d_in[11];
  const float* bih1  = (const float*)d_in[12];
  const float* bhh1  = (const float*)d_in[13];
  const float* wih2  = (const float*)d_in[14];
  /* d_in[15] whh2 unused: second LSTM runs from zero hidden state */
  const float* bih2  = (const float*)d_in[16];
  const float* bhh2  = (const float*)d_in[17];
  const float* wf    = (const float*)d_in[18];
  const float* bfv   = (const float*)d_in[19];
  const float* lam   = (const float*)d_in[20];
  float* out = (float*)d_out;

  char* ws = (char*)d_ws;
  half_t* w1t   = (half_t*)(ws + OFF_W1T);
  half_t* w2t   = (half_t*)(ws + OFF_W2T);
  half_t* Wc1   = (half_t*)(ws + OFF_WC1);
  half_t* wih2t = (half_t*)(ws + OFF_WIH2T);
  float*  bc1   = (float*)(ws + OFF_BC1);
  float*  bc2   = (float*)(ws + OFF_BC2);
  float*  dinv  = (float*)(ws + OFF_DINV);
  half_t* XW1T  = (half_t*)(ws + OFF_XW1T);
  half_t* Hf    = (half_t*)(ws + OFF_HF);
  half_t* HW2T  = (half_t*)(ws + OFF_HW2T);
  half_t* Z     = (half_t*)(ws + OFF_Z);
  half_t* H1    = (half_t*)(ws + OFF_H1);
  float*  C1    = (float*)(ws + OFF_C1);
  float*  H2    = (float*)(ws + OFF_H2);

  prep_weights_kernel<<<1, 256, 0, stream>>>(w1, w2, wih1, whh1, bih1, bhh1, wih2, bih2, bhh2,
                                             w1t, w2t, Wc1, wih2t, bc1, bc2);
  zero_state_kernel<<<(NODES * 128 + 255) / 256, 256, 0, stream>>>(H1, C1);
  degree_kernel<<<BD * Nn / 8, 256, 0, stream>>>(A, dinv);
  xw1_kernel<<<BD * 8, 256, 0, stream>>>(X, w1t, XW1T);
  gcn_kernel<HID, true><<<BD * 8, 256, 0, stream>>>(A, dinv, XW1T, b1, Hf, HID);
  hw2_kernel<<<BD * 8, 256, 0, stream>>>(Hf, w2t, HW2T);
  gcn_kernel<OUTC, false><<<BD * 8, 256, 0, stream>>>(A, dinv, HW2T, b2, Z, 128);
  text_kernel<<<(int)(((size_t)BD * Nn * OUTC + 255) / 256), 256, 0, stream>>>(extra, idxv, wt, bt, lam, Z);
  for (int t = 0; t < Dd; ++t)
    lstm1_step_kernel<<<NODES / 16, 256, 0, stream>>>(Z, Wc1, bc1, H1, C1, t);
  lstm2_kernel<<<NODES / 32, 256, 0, stream>>>(H1, wih2t, bc2, H2);
  final_kernel<<<(NODES * Yn + 255) / 256, 256, 0, stream>>>(H2, wf, bfv, out);
}
